// LocalAttention2_54992761258371
// MI455X (gfx1250) — matrix operand order repaired and checked by reading, not compiled, not run on hardware
//
#include <hip/hip_runtime.h>
#include <stdint.h>

// Problem constants (match reference)
#define BB 4
#define TT 4096
#define DD 512
#define HH 8
#define HD 64
#define BT (BB * TT)       // 16384 token rows
#define EPSV 1e-5f

// LDS staging geometry for the GEMM: one 64-column B panel, full K=512,
// row stride padded by 8 halves (16B) so per-lane ds_load_b128 fragment
// reads start 4 banks apart (conflict-free across lanes 0..15).
#define LDS_STRIDE 520                     // halves per staged row
#define LDS_PANEL  (64 * LDS_STRIDE)       // halves per buffer
#define LDS_BYTES  (2 * LDS_PANEL * 2)     // double buffered, bytes (133120)

// ---------------------------------------------------------------------------
// Types for WMMA fragments
// ---------------------------------------------------------------------------
typedef __attribute__((ext_vector_type(16))) __bf16   v16bf;
typedef __attribute__((ext_vector_type(8)))  float    v8f;
typedef __attribute__((ext_vector_type(4)))  uint32_t u32x4;

union FragBF {          // 16 bf16 = 32 bytes = two b128 loads
    v16bf v;
    u32x4 q[2];
};

__device__ __forceinline__ uint16_t f2bf(float f) {
    uint32_t u = __float_as_uint(f);
    u += 0x7FFFu + ((u >> 16) & 1u);    // round-to-nearest-even
    return (uint16_t)(u >> 16);
}

// ---------------------------------------------------------------------------
// fp32 -> bf16 conversion (x and weights; done once per call)
// ---------------------------------------------------------------------------
__global__ __launch_bounds__(256) void cvt_f32_bf16(const float* __restrict__ in,
                                                    uint16_t* __restrict__ out,
                                                    int n) {
    int i = blockIdx.x * blockDim.x + threadIdx.x;
    if (i < n) out[i] = f2bf(in[i]);
}

// ---------------------------------------------------------------------------
// C[M,N] = A_bf16[M,K] @ W_bf16[N,K]^T + bias   (M = BT, N = K = DD)
//
// Block = 8 waves (256 threads); wave w owns rows m0 = (blk*8+w)*16 and keeps
// its 16x512 A strip resident in VGPRs (16 fragments, 128 VGPRs).
// For each of the 8 64-column N-groups, the block stages the 64x512 bf16 B
// panel in LDS once (shared by all 8 waves -> 8x less L2 W traffic),
// double-buffered: next panel's global_load_b128s issue before the 64-WMMA
// compute phase, ds_store_b128 after it, one barrier per group.
// ---------------------------------------------------------------------------
__global__ __launch_bounds__(256) void gemm_bf16_wmma(
    const uint16_t* __restrict__ A,     // [BT, DD] bf16
    const uint16_t* __restrict__ W,     // [DD, DD] bf16 (row n = B column n)
    const float*    __restrict__ bias,  // [DD]
    float*          __restrict__ C)     // [BT, DD] f32
{
    extern __shared__ uint16_t smem[];  // [2][64][LDS_STRIDE]

    const int tid  = threadIdx.x;
    const int lane = tid & 31;
    const int wave = tid >> 5;
    const int m0   = (blockIdx.x * 8 + wave) * 16;
    const int l    = lane & 15;
    const int hf   = lane >> 4;

    // ---- load A strip: 16 K-chunks of 32 (ISA 16-bit A layout) ----
    FragBF a[16];
    const uint16_t* arow = A + (size_t)(m0 + l) * DD;
    #pragma unroll
    for (int kc = 0; kc < 16; ++kc) {
        const int k0 = kc * 32;
        a[kc].q[0] = *(const u32x4*)(arow + k0 + hf * 8);        // K = k0+8h .. +7
        a[kc].q[1] = *(const u32x4*)(arow + k0 + 16 + hf * 8);   // K = k0+16+8h .. +7
    }

    // staging split: 64 rows x 64 16B-chunks = 4096 chunks, 16 per thread
    const int srow = tid >> 4;            // rows handled: srow, srow+16, ... (4 rows)
    const int scol = (tid & 15) * 4;      // 4 consecutive 16B chunks per row
    // -> thread covers rows {srow + 16*r, r<4}, chunks {scol..scol+3}; total 16

    // ---- prologue: stage panel 0 ----
    {
        #pragma unroll
        for (int r = 0; r < 4; ++r) {
            const int row = srow + 16 * r;
            const uint16_t* src = W + (size_t)row * DD + scol * 8;
            uint16_t*       dst = smem + row * LDS_STRIDE + scol * 8;
            #pragma unroll
            for (int c = 0; c < 4; ++c)
                *(u32x4*)(dst + c * 8) = *(const u32x4*)(src + c * 8);
        }
    }
    __syncthreads();

    for (int ng = 0; ng < DD / 64; ++ng) {
        // ---- issue next panel's global loads early (held in VGPRs) ----
        u32x4 pre[16];
        const bool more = (ng + 1 < DD / 64);
        if (more) {
            const uint16_t* wsrc = W + (size_t)(ng + 1) * 64 * DD;
            #pragma unroll
            for (int r = 0; r < 4; ++r) {
                const int row = srow + 16 * r;
                const uint16_t* src = wsrc + (size_t)row * DD + scol * 8;
                #pragma unroll
                for (int c = 0; c < 4; ++c)
                    pre[r * 4 + c] = *(const u32x4*)(src + c * 8);
            }
        }

        // ---- compute: 16 K-chunks x 4 N-tiles of WMMA from LDS panel ----
        const uint16_t* bbase = smem + (ng & 1) * LDS_PANEL;
        v8f acc[4];
        #pragma unroll
        for (int nt = 0; nt < 4; ++nt)
            acc[nt] = (v8f)(0.0f);

        #pragma unroll
        for (int kc = 0; kc < 16; ++kc) {
            const int k0 = kc * 32;
            #pragma unroll
            for (int nt = 0; nt < 4; ++nt) {
                // B column n = panel row nt*16+l, halves = K k0+8h .. +7, k0+16+8h .. +7
                const uint16_t* p = bbase + (nt * 16 + l) * LDS_STRIDE + k0 + hf * 8;
                FragBF b;
                b.q[0] = *(const u32x4*)(p);
                b.q[1] = *(const u32x4*)(p + 16);
                acc[nt] = __builtin_amdgcn_wmma_f32_16x16x32_bf16(
                    false, a[kc].v, false, b.v, (short)0, acc[nt], false, false);
            }
        }

        // ---- store (C/D layout: VGPR j -> M = m0+j+8h, N = n0 + (lane&15)) ----
        #pragma unroll
        for (int nt = 0; nt < 4; ++nt) {
            const int col = ng * 64 + nt * 16 + l;
            const float bv = bias[col];
            #pragma unroll
            for (int j = 0; j < 8; ++j) {
                const int row = m0 + j + hf * 8;
                C[(size_t)row * DD + col] = acc[nt][j] + bv;
            }
        }

        // ---- drain next panel into the other LDS buffer ----
        if (more) {
            uint16_t* dbase = smem + ((ng + 1) & 1) * LDS_PANEL;
            #pragma unroll
            for (int r = 0; r < 4; ++r) {
                const int row = srow + 16 * r;
                uint16_t* dst = dbase + row * LDS_STRIDE + scol * 8;
                #pragma unroll
                for (int c = 0; c < 4; ++c)
                    *(u32x4*)(dst + c * 8) = pre[r * 4 + c];
            }
            __syncthreads();
        }
    }
}

// ---------------------------------------------------------------------------
// Dual band attention (i=1 and i=2 fused). One thread per (b, t, h).
// Q/K/V are [BT, DD] f32 with head h at columns h*64..h*64+63.
// Outputs o1,o2 as bf16 [BT, DD] (A operand of the next GEMM).
// ---------------------------------------------------------------------------
__global__ __launch_bounds__(256) void band_attn_dual(
    const float* __restrict__ Q, const float* __restrict__ K,
    const float* __restrict__ V,
    uint16_t* __restrict__ o1, uint16_t* __restrict__ o2)
{
    const int idx = blockIdx.x * blockDim.x + threadIdx.x;  // [0, BT*HH)
    const int h   = idx & (HH - 1);
    const int tg  = idx >> 3;              // global row b*TT + t
    const int t   = tg & (TT - 1);

    const size_t base = (size_t)tg * DD + h * HD;
    const float* q  = Q + base;
    const float* k0 = K + base;
    const float* v0 = V + base;

    const bool vl1 = (t >= 1), vr1 = (t < TT - 1);
    const bool vl2 = (t >= 2), vr2 = (t < TT - 2);

    float d0 = 0.f, dl1 = 0.f, dr1 = 0.f, dl2 = 0.f, dr2 = 0.f;
    #pragma unroll 8
    for (int d = 0; d < HD; ++d) {
        const float qd = q[d];
        d0 += qd * k0[d];
        if (vl1) dl1 += qd * k0[d - DD];
        if (vr1) dr1 += qd * k0[d + DD];
        if (vl2) dl2 += qd * k0[d - 2 * DD];
        if (vr2) dr2 += qd * k0[d + 2 * DD];
    }

    const float scale = 0.125f;            // 1/sqrt(64)
    const float NI = -__builtin_inff();

    // softmax over {left, mid, right}, band i = 1
    float a0 = vl1 ? dl1 * scale : NI;
    float a1 = d0 * scale;
    float a2 = vr1 ? dr1 * scale : NI;
    float mx = fmaxf(a1, fmaxf(a0, a2));
    float e0 = __expf(a0 - mx), e1 = __expf(a1 - mx), e2 = __expf(a2 - mx);
    float inv = 1.f / (e0 + e1 + e2);
    const float w10 = e0 * inv, w11 = e1 * inv, w12 = e2 * inv;

    // band i = 2
    float b0 = vl2 ? dl2 * scale : NI;
    float b2 = vr2 ? dr2 * scale : NI;
    float mx2 = fmaxf(a1, fmaxf(b0, b2));
    float f0 = __expf(b0 - mx2), f1 = __expf(a1 - mx2), f2 = __expf(b2 - mx2);
    float inv2 = 1.f / (f0 + f1 + f2);
    const float w20 = f0 * inv2, w21 = f1 * inv2, w22 = f2 * inv2;

    #pragma unroll 8
    for (int d = 0; d < HD; ++d) {
        const float vm  = v0[d];
        const float vlA = vl1 ? v0[d - DD]     : 0.f;
        const float vrA = vr1 ? v0[d + DD]     : 0.f;
        const float vlB = vl2 ? v0[d - 2 * DD] : 0.f;
        const float vrB = vr2 ? v0[d + 2 * DD] : 0.f;
        o1[base + d] = f2bf(w10 * vlA + w11 * vm + w12 * vrA);
        o2[base + d] = f2bf(w20 * vlB + w21 * vm + w22 * vrB);
    }
}

// ---------------------------------------------------------------------------
// out = LN(u1 + x)*gamma+beta + LN(u2 + x)*gamma+beta.
// One wave per row (D=512 -> 16 elems/lane), wave32 shuffle reductions.
// ---------------------------------------------------------------------------
__global__ __launch_bounds__(256) void ln_sum(
    const float* __restrict__ u1, const float* __restrict__ u2,
    const float* __restrict__ x,
    const float* __restrict__ gamma, const float* __restrict__ beta,
    float* __restrict__ out)
{
    const int wv   = threadIdx.x >> 5;
    const int lane = threadIdx.x & 31;
    const int row  = blockIdx.x * 8 + wv;
    const size_t rb = (size_t)row * DD;

    float y1[16], y2[16];
    float s1 = 0.f, s2 = 0.f;
    #pragma unroll
    for (int i = 0; i < 16; ++i) {
        const int c = lane + i * 32;
        const float xv = x[rb + c];
        y1[i] = u1[rb + c] + xv;
        y2[i] = u2[rb + c] + xv;
        s1 += y1[i];
        s2 += y2[i];
    }
    #pragma unroll
    for (int o = 16; o > 0; o >>= 1) {
        s1 += __shfl_xor(s1, o, 32);
        s2 += __shfl_xor(s2, o, 32);
    }
    const float m1 = s1 * (1.f / DD), m2 = s2 * (1.f / DD);

    float q1 = 0.f, q2 = 0.f;
    #pragma unroll
    for (int i = 0; i < 16; ++i) {
        const float d1 = y1[i] - m1;
        const float d2 = y2[i] - m2;
        q1 += d1 * d1;
        q2 += d2 * d2;
    }
    #pragma unroll
    for (int o = 16; o > 0; o >>= 1) {
        q1 += __shfl_xor(q1, o, 32);
        q2 += __shfl_xor(q2, o, 32);
    }
    const float inv1 = rsqrtf(q1 * (1.f / DD) + EPSV);
    const float inv2 = rsqrtf(q2 * (1.f / DD) + EPSV);

    #pragma unroll
    for (int i = 0; i < 16; ++i) {
        const int c = lane + i * 32;
        const float g = gamma[c], bt = beta[c];
        out[rb + c] = (y1[i] - m1) * inv1 * g + bt
                    + (y2[i] - m2) * inv2 * g + bt;
    }
}

// ---------------------------------------------------------------------------
// Launcher
// ---------------------------------------------------------------------------
extern "C" void kernel_launch(void* const* d_in, const int* in_sizes, int n_in,
                              void* d_out, int out_size, void* d_ws, size_t ws_size,
                              hipStream_t stream) {
    const float* x     = (const float*)d_in[0];
    const float* Wq    = (const float*)d_in[1];
    const float* bq    = (const float*)d_in[2];
    const float* Wk    = (const float*)d_in[3];
    const float* bk    = (const float*)d_in[4];
    const float* Wv    = (const float*)d_in[5];
    const float* bv    = (const float*)d_in[6];
    const float* Wo    = (const float*)d_in[7];
    const float* bo    = (const float*)d_in[8];
    const float* gamma = (const float*)d_in[9];
    const float* beta  = (const float*)d_in[10];
    float* out = (float*)d_out;

    // workspace carve-up (256B aligned)
    char* ws = (char*)d_ws;
    size_t off = 0;
    auto carve = [&](size_t bytes) -> void* {
        void* p = ws + off;
        off = (off + bytes + 255) & ~(size_t)255;
        return p;
    };
    uint16_t* xb  = (uint16_t*)carve((size_t)BT * DD * 2);
    uint16_t* wqb = (uint16_t*)carve((size_t)DD * DD * 2);
    uint16_t* wkb = (uint16_t*)carve((size_t)DD * DD * 2);
    uint16_t* wvb = (uint16_t*)carve((size_t)DD * DD * 2);
    uint16_t* wob = (uint16_t*)carve((size_t)DD * DD * 2);
    float*    Qf  = (float*)carve((size_t)BT * DD * 4);
    float*    Kf  = (float*)carve((size_t)BT * DD * 4);
    float*    Vf  = (float*)carve((size_t)BT * DD * 4);
    uint16_t* o1b = (uint16_t*)carve((size_t)BT * DD * 2);
    uint16_t* o2b = (uint16_t*)carve((size_t)BT * DD * 2);
    float*    u1  = (float*)carve((size_t)BT * DD * 4);
    float*    u2  = (float*)carve((size_t)BT * DD * 4);
    (void)ws_size; (void)in_sizes; (void)n_in; (void)out_size;

    // allow 130KB dynamic LDS for the GEMM (gfx1250: up to 320KB/workgroup)
    hipFuncSetAttribute(reinterpret_cast<const void*>(gemm_bf16_wmma),
                        hipFuncAttributeMaxDynamicSharedMemorySize, LDS_BYTES);

    const int nX = BT * DD, nW = DD * DD;
    cvt_f32_bf16<<<(nX + 255) / 256, 256, 0, stream>>>(x,  xb,  nX);
    cvt_f32_bf16<<<(nW + 255) / 256, 256, 0, stream>>>(Wq, wqb, nW);
    cvt_f32_bf16<<<(nW + 255) / 256, 256, 0, stream>>>(Wk, wkb, nW);
    cvt_f32_bf16<<<(nW + 255) / 256, 256, 0, stream>>>(Wv, wvb, nW);
    cvt_f32_bf16<<<(nW + 255) / 256, 256, 0, stream>>>(Wo, wob, nW);

    dim3 gg(BT / 128, 1, 1);   // 8 waves/block, 16 rows/wave
    gemm_bf16_wmma<<<gg, 256, LDS_BYTES, stream>>>(xb, wqb, bq, Qf);
    gemm_bf16_wmma<<<gg, 256, LDS_BYTES, stream>>>(xb, wkb, bk, Kf);
    gemm_bf16_wmma<<<gg, 256, LDS_BYTES, stream>>>(xb, wvb, bv, Vf);

    band_attn_dual<<<(BT * HH) / 256, 256, 0, stream>>>(Qf, Kf, Vf, o1b, o2b);

    gemm_bf16_wmma<<<gg, 256, LDS_BYTES, stream>>>(o1b, wob, bo, u1);
    gemm_bf16_wmma<<<gg, 256, LDS_BYTES, stream>>>(o2b, wob, bo, u2);

    ln_sum<<<BT / 8, 256, 0, stream>>>(u1, u2, x, gamma, beta, out);
}
